// DeepSeekMoE_63909113364650
// MI455X (gfx1250) — compile-verified, matrix operand order checked by reading, not run on hardware
//
#include <hip/hip_runtime.h>
#include <hip/hip_bf16.h>
#include <math.h>

// ---------------- problem constants ----------------
#define HIDDEN 1024
#define INTER  512
#define NEXP   64
#define TOPK   6
#define NGRP   8
#define TOPG   4
#define ISHX   1024   // shared-expert intermediate (2 * 512)
#define TOK    1024
#define CAP    1024   // worst-case tokens per expert

typedef __attribute__((ext_vector_type(16))) __bf16 v16bf;
typedef __attribute__((ext_vector_type(8)))  float  v8f;

union FragV { v16bf v; unsigned short h[16]; unsigned int u[8]; };

// pack two fp32 (bit patterns) into two bf16 (truncate) in ONE v_perm_b32:
// result = [hi[31:16] , lo[31:16]]
__device__ __forceinline__ unsigned int pk2_trunc(unsigned ulo, unsigned uhi) {
#if __has_builtin(__builtin_amdgcn_perm)
  return __builtin_amdgcn_perm(uhi, ulo, 0x07060302u);
#else
  return (ulo >> 16) | (uhi & 0xFFFF0000u);
#endif
}

// single fp32 -> bf16, round-to-nearest (ties away); epilogue only (not in k-loop)
__device__ __forceinline__ unsigned short f2bf_rn(float v) {
  return (unsigned short)((__float_as_uint(v) + 0x8000u) >> 16);
}

__device__ __forceinline__ float fast_rcp(float x) {
#if __has_builtin(__builtin_amdgcn_rcpf)
  return __builtin_amdgcn_rcpf(x);
#else
  return 1.f / x;
#endif
}

// 16x32 bf16 fragment from row-major fp32 [rows x ld] (ld, k0 multiples of 32 -> 16B aligned).
// Lane layout (ISA 7.12.2): M = lane&15; lanes>=16 take K+8; elems 0..7 -> K, 8..15 -> K+16.
__device__ __forceinline__ v16bf frag_ld_f32(const float* __restrict__ p, int ld, int row0, int k0) {
  const int lane = threadIdx.x & 31;
  const float* s = p + (size_t)(row0 + (lane & 15)) * ld + k0 + ((lane >> 4) << 3);
  uint4 a0 = *(const uint4*)(s);
  uint4 a1 = *(const uint4*)(s + 4);
  uint4 b0 = *(const uint4*)(s + 16);
  uint4 b1 = *(const uint4*)(s + 20);
  FragV f;
  f.u[0] = pk2_trunc(a0.x, a0.y);
  f.u[1] = pk2_trunc(a0.z, a0.w);
  f.u[2] = pk2_trunc(a1.x, a1.y);
  f.u[3] = pk2_trunc(a1.z, a1.w);
  f.u[4] = pk2_trunc(b0.x, b0.y);
  f.u[5] = pk2_trunc(b0.z, b0.w);
  f.u[6] = pk2_trunc(b1.x, b1.y);
  f.u[7] = pk2_trunc(b1.z, b1.w);
  return f.v;
}

// fragment from bf16 (ushort) memory (global or LDS), same layout; 16B-aligned runs
__device__ __forceinline__ v16bf frag_ld_bf16(const unsigned short* __restrict__ p, int ld, int row0, int k0) {
  const int lane = threadIdx.x & 31;
  const unsigned short* s = p + (size_t)(row0 + (lane & 15)) * ld + k0 + ((lane >> 4) << 3);
  FragV f;
  *(uint4*)&f.u[0] = *(const uint4*)(s);
  *(uint4*)&f.u[4] = *(const uint4*)(s + 16);
  return f.v;
}

__device__ __forceinline__ v8f wmma_bf16(v16bf a, v16bf b, v8f c) {
  return __builtin_amdgcn_wmma_f32_16x16x32_bf16(false, a, false, b, (short)0, c, false, false);
}

// ---------------- kernel 1: gate + routing ----------------
__global__ __launch_bounds__(128) void moe_gate_kernel(
    const float* __restrict__ x, const float* __restrict__ gw, const float* __restrict__ gb,
    int* __restrict__ cnt, int* __restrict__ idx, float* __restrict__ wt) {
  const int t = blockIdx.x;
  __shared__ float xs[HIDDEN];
  __shared__ float sc[NEXP];
  for (int i = threadIdx.x; i < HIDDEN; i += blockDim.x) xs[i] = x[(size_t)t * HIDDEN + i];
  __syncthreads();
  if (threadIdx.x < NEXP) {
    const float* w = gw + (size_t)threadIdx.x * HIDDEN;
    float acc = 0.f;
    for (int h = 0; h < HIDDEN; ++h) acc = fmaf(xs[h], w[h], acc);
    sc[threadIdx.x] = fast_rcp(1.f + __expf(-acc)) + gb[threadIdx.x];
  }
  __syncthreads();
  if (threadIdx.x == 0) {
    float gs[NGRP];
#pragma unroll
    for (int g = 0; g < NGRP; ++g) {
      float m = sc[g * 8];
#pragma unroll
      for (int j = 1; j < 8; ++j) m = fmaxf(m, sc[g * 8 + j]);
      gs[g] = m;
    }
    bool gsel[NGRP] = {};
#pragma unroll
    for (int it = 0; it < TOPG; ++it) {
      int best = 0; float bv = -1e30f;
#pragma unroll
      for (int g = 0; g < NGRP; ++g)
        if (!gsel[g] && gs[g] > bv) { bv = gs[g]; best = g; }
      gsel[best] = true;
    }
    float ms[NEXP];
#pragma unroll
    for (int e = 0; e < NEXP; ++e) ms[e] = gsel[e >> 3] ? sc[e] : 0.f;
    int ti[TOPK]; float tv[TOPK]; float s = 1e-8f;
    bool used[NEXP] = {};
#pragma unroll
    for (int k = 0; k < TOPK; ++k) {
      int best = 0; float bv = -1e30f;
      for (int e = 0; e < NEXP; ++e)
        if (!used[e] && ms[e] > bv) { bv = ms[e]; best = e; }
      used[best] = true; ti[k] = best; tv[k] = bv; s += bv;
    }
    const float inv = 1.f / s;
#pragma unroll
    for (int k = 0; k < TOPK; ++k) {
      const int e = ti[k];
      const int p = atomicAdd(&cnt[e], 1);
      idx[e * CAP + p] = t;
      wt[e * CAP + p]  = tv[k] * inv;
    }
  }
}

// ---------------- kernel 2: shared expert gate/up (hs = silu(x sg^T) * (x su^T), bf16) ----------------
#define XPAD 8
__global__ __launch_bounds__(256) void moe_shared_gateup_kernel(
    const float* __restrict__ x, const float* __restrict__ sg, const float* __restrict__ su,
    unsigned short* __restrict__ hsh) {
  const int wbase = blockIdx.x * 8;
  const int t0 = (wbase >> 6) * 16;                  // 8 waves of a block share one A tile
  __shared__ unsigned short xls[16 * (HIDDEN + XPAD)];
  for (int p = threadIdx.x; p < 16 * (HIDDEN / 2); p += blockDim.x) {
    const int m = p >> 9;                            // HIDDEN/2 = 512 pairs per row
    const int kk = (p & 511) << 1;
    const uint2 v = *(const uint2*)(x + (size_t)(t0 + m) * HIDDEN + kk);
    *(unsigned int*)&xls[m * (HIDDEN + XPAD) + kk] = pk2_trunc(v.x, v.y);
  }
  __syncthreads();

  const int gw_ = wbase + (threadIdx.x >> 5);
  const int n0 = (gw_ & 63) * 16;
  v8f cg = {}, cu = {};
#pragma unroll 2
  for (int k0 = 0; k0 < HIDDEN; k0 += 32) {
    v16bf a  = frag_ld_bf16(xls, HIDDEN + XPAD, 0, k0);
    v16bf bg = frag_ld_f32(sg, HIDDEN, n0, k0);
    v16bf bu = frag_ld_f32(su, HIDDEN, n0, k0);
    cg = wmma_bf16(a, bg, cg);
    cu = wmma_bf16(a, bu, cu);
  }
  const int lane = threadIdx.x & 31;
  const int half8 = (lane >> 4) << 3;
  const int col = lane & 15;
#pragma unroll
  for (int r = 0; r < 8; ++r) {
    const float g = cg[r], u = cu[r];
    const float v = g * fast_rcp(1.f + __expf(-g)) * u;   // silu(g) * u
    hsh[(size_t)(t0 + r + half8) * ISHX + n0 + col] = f2bf_rn(v);
  }
}

// ---------------- kernel 3: shared expert down (out = hs @ sd^T, plain store) ----------------
__global__ __launch_bounds__(256) void moe_shared_down_kernel(
    const unsigned short* __restrict__ hsh, const float* __restrict__ sd,
    float* __restrict__ out) {
  const int wbase = blockIdx.x * 8;
  const int t0 = (wbase >> 6) * 16;
  __shared__ unsigned short xls[16 * (ISHX + XPAD)];
  for (int p = threadIdx.x; p < 16 * (ISHX / 2); p += blockDim.x) {
    const int m = p >> 9;
    const int kk = (p & 511) << 1;
    *(unsigned int*)&xls[m * (ISHX + XPAD) + kk] =
        *(const unsigned int*)(hsh + (size_t)(t0 + m) * ISHX + kk);
  }
  __syncthreads();

  const int gw_ = wbase + (threadIdx.x >> 5);
  const int n0 = (gw_ & 63) * 16;
  v8f c = {};
#pragma unroll 2
  for (int k0 = 0; k0 < ISHX; k0 += 32) {
    v16bf a = frag_ld_bf16(xls, ISHX + XPAD, 0, k0);
    v16bf b = frag_ld_f32(sd, ISHX, n0, k0);
    c = wmma_bf16(a, b, c);
  }
  const int lane = threadIdx.x & 31;
  const int half8 = (lane >> 4) << 3;
  const int col = lane & 15;
#pragma unroll
  for (int r = 0; r < 8; ++r)
    out[(size_t)(t0 + r + half8) * HIDDEN + n0 + col] = c[r];
}

// ---------------- kernel 4: routed experts (sparse, fused SwiGLU + down-proj) ----------------
// Each wave carries TWO 16-row M-tiles so every weight (B) fragment is loaded
// and converted exactly once per column tile.
#define MB   32     // tokens per block tile
#define IPAD 8      // LDS stride padding (shorts)

__global__ __launch_bounds__(256) void moe_routed_kernel(
    const float* __restrict__ x,  const float* __restrict__ eg,
    const float* __restrict__ eu, const float* __restrict__ ed,
    const int* __restrict__ cnt,  const int* __restrict__ idx,
    const float* __restrict__ wt, float* __restrict__ out) {
  const int e  = blockIdx.x >> 5;          // 32 token-tiles max per expert
  const int m0 = (blockIdx.x & 31) * MB;
  const int n  = cnt[e];
  if (m0 >= n) return;

  __shared__ unsigned short xls[MB * (HIDDEN + XPAD)];  // gathered X tile, bf16 (66 KB)
  __shared__ unsigned short hs[MB * (INTER + IPAD)];    // weighted intermediate (33 KB)
  __shared__ int   tok[MB];
  __shared__ float twt[MB];
  for (int i = threadIdx.x; i < MB; i += blockDim.x) {
    const int gi = m0 + i;
    tok[i] = (gi < n) ? idx[e * CAP + gi] : -1;
    twt[i] = (gi < n) ? wt[e * CAP + gi]  : 0.f;
  }
  __syncthreads();

  // Phase 0: gather+convert token rows into LDS (once, instead of per column tile)
  for (int p = threadIdx.x; p < MB * (HIDDEN / 2); p += blockDim.x) {
    const int m = p >> 9;
    const int kk = (p & 511) << 1;
    const int t = tok[m];
    uint2 v = make_uint2(0u, 0u);
    if (t >= 0) v = *(const uint2*)(x + (size_t)t * HIDDEN + kk);
    *(unsigned int*)&xls[m * (HIDDEN + XPAD) + kk] = pk2_trunc(v.x, v.y);
  }
  __syncthreads();

  const int wave = threadIdx.x >> 5;
  const int lane = threadIdx.x & 31;
  const int half8 = (lane >> 4) << 3;
  const int col = lane & 15;
  const float* G = eg + (size_t)e * INTER * HIDDEN;
  const float* U = eu + (size_t)e * INTER * HIDDEN;
  const float* D = ed + (size_t)e * HIDDEN * INTER;

  // Phase 1: hs[MB, INTER] = route_w * silu(Xe G^T) * (Xe U^T)
  // 32 column tiles over 8 waves; both M-tiles per wave -> B loaded once.
  for (int i0 = wave * 16; i0 < INTER; i0 += 8 * 16) {
    v8f cg0 = {}, cu0 = {}, cg1 = {}, cu1 = {};
#pragma unroll 2
    for (int k0 = 0; k0 < HIDDEN; k0 += 32) {
      v16bf a0 = frag_ld_bf16(xls, HIDDEN + XPAD, 0,  k0);
      v16bf a1 = frag_ld_bf16(xls, HIDDEN + XPAD, 16, k0);
      v16bf bg = frag_ld_f32(G, HIDDEN, i0, k0);
      v16bf bu = frag_ld_f32(U, HIDDEN, i0, k0);
      cg0 = wmma_bf16(a0, bg, cg0);
      cu0 = wmma_bf16(a0, bu, cu0);
      cg1 = wmma_bf16(a1, bg, cg1);
      cu1 = wmma_bf16(a1, bu, cu1);
    }
#pragma unroll
    for (int r = 0; r < 8; ++r) {
      const int m = r + half8;
      const float g0 = cg0[r], u0 = cu0[r];
      hs[m * (INTER + IPAD) + i0 + col] =
          f2bf_rn(g0 * fast_rcp(1.f + __expf(-g0)) * u0 * twt[m]);
      const int m1 = m + 16;
      const float g1 = cg1[r], u1 = cu1[r];
      hs[m1 * (INTER + IPAD) + i0 + col] =
          f2bf_rn(g1 * fast_rcp(1.f + __expf(-g1)) * u1 * twt[m1]);
    }
  }
  __syncthreads();

  // Phase 2: out[tok, :] += hs @ D^T  (atomic combine across experts)
  // 64 column tiles over 8 waves; both M-tiles per wave -> B loaded once.
  for (int h0 = wave * 16; h0 < HIDDEN; h0 += 8 * 16) {
    v8f c0 = {}, c1 = {};
#pragma unroll 2
    for (int k0 = 0; k0 < INTER; k0 += 32) {
      v16bf a0 = frag_ld_bf16(hs, INTER + IPAD, 0,  k0);
      v16bf a1 = frag_ld_bf16(hs, INTER + IPAD, 16, k0);
      v16bf b  = frag_ld_f32(D, INTER, h0, k0);
      c0 = wmma_bf16(a0, b, c0);
      c1 = wmma_bf16(a1, b, c1);
    }
#pragma unroll
    for (int r = 0; r < 8; ++r) {
      const int m = r + half8;
      const int t0_ = tok[m];
      if (t0_ >= 0) atomicAdd(&out[(size_t)t0_ * HIDDEN + h0 + col], c0[r]);
      const int t1_ = tok[m + 16];
      if (t1_ >= 0) atomicAdd(&out[(size_t)t1_ * HIDDEN + h0 + col], c1[r]);
    }
  }
}

// ---------------- launch ----------------
extern "C" void kernel_launch(void* const* d_in, const int* in_sizes, int n_in,
                              void* d_out, int out_size, void* d_ws, size_t ws_size,
                              hipStream_t stream) {
  const float* x  = (const float*)d_in[0];
  const float* gw = (const float*)d_in[1];
  const float* gb = (const float*)d_in[2];
  const float* eg = (const float*)d_in[3];
  const float* eu = (const float*)d_in[4];
  const float* ed = (const float*)d_in[5];
  const float* sg = (const float*)d_in[6];
  const float* su = (const float*)d_in[7];
  const float* sd = (const float*)d_in[8];
  float* out = (float*)d_out;

  char* ws = (char*)d_ws;
  int*   cnt = (int*)ws;                                            // 64 ints
  int*   idx = (int*)(ws + 1024);                                   // 256 KB
  float* wtb = (float*)(ws + 1024 + (size_t)NEXP * CAP * 4);        // 256 KB
  unsigned short* hsh = (unsigned short*)(ws + 1024 + 2 * (size_t)NEXP * CAP * 4); // 2 MB bf16

  hipMemsetAsync(cnt, 0, NEXP * sizeof(int), stream);
  moe_gate_kernel<<<TOK, 128, 0, stream>>>(x, gw, gb, cnt, idx, wtb);
  moe_shared_gateup_kernel<<<512, 256, 0, stream>>>(x, sg, su, hsh);
  moe_shared_down_kernel<<<512, 256, 0, stream>>>(hsh, sd, out);
  moe_routed_kernel<<<NEXP * 32, 256, 0, stream>>>(x, eg, eu, ed, cnt, idx, wtb, out);
}
